// MultipleSequenceHybridMamba_47416438948488
// MI455X (gfx1250) — compile-verified, matrix operand order checked by reading, not run on hardware
//
#include <hip/hip_runtime.h>
#include <math.h>

// ---------------- model dims ----------------
#define BB 2
#define LL 2048
#define BL (BB*LL)          // 4096
#define DM 256
#define DI 512
#define DS 128
#define NH 8
#define HD 64
#define CONVD 768           // DI + 2*DS
#define DIP 1288            // 2*DI + 2*DS + NH
#define MLPH 1024
#define AH 8
#define AHD 32

typedef __attribute__((ext_vector_type(16))) __bf16 v16bf;
typedef __attribute__((ext_vector_type(8)))  float  v8f;

__device__ __forceinline__ float siluf(float x) { return x / (1.f + __expf(-x)); }
__device__ __forceinline__ float geluf(float x) { return 0.5f * x * (1.f + erff(x * 0.70710678118654752f)); }

__device__ __forceinline__ v8f wmma_bf16(v16bf a, v16bf b, v8f c) {
    return __builtin_amdgcn_wmma_f32_16x16x32_bf16(false, a, false, b, (short)0, c, false, false);
}

// ---------------- cast f32 -> bf16 ----------------
__global__ void cast_bf16_kernel(const float* __restrict__ src, __bf16* __restrict__ dst, int n) {
    int i = blockIdx.x * blockDim.x + threadIdx.x;
    if (i < n) dst[i] = (__bf16)src[i];
}

// ---------------- generic WMMA GEMM: out[M,N] = A[M,K] @ W[N,K]^T (+bias)(+resid)(act) ----
// 256 threads = 8 waves; block tile 128(M) x 64(N); each wave does 32x32 via 4 WMMA accs.
__global__ void gemm_wmma_kernel(const __bf16* __restrict__ A, int lda,
                                 const __bf16* __restrict__ W, int ldw,
                                 float* __restrict__ outF, __bf16* __restrict__ outB,
                                 int ldc, int M, int N, int K,
                                 const float* __restrict__ bias,
                                 const float* __restrict__ resid,
                                 int act, int transposed, int Lseq) {
    const int w    = threadIdx.x >> 5;     // wave32
    const int lane = threadIdx.x & 31;
    const int half = lane >> 4;
    const int r16  = lane & 15;
    const int m0 = blockIdx.x * 128 + (w & 3) * 32;
    const int n0 = blockIdx.y * 64  + (w >> 2) * 32;

    const int ar0 = min(m0 + r16,      M - 1);
    const int ar1 = min(m0 + 16 + r16, M - 1);
    const int br0 = min(n0 + r16,      N - 1);
    const int br1 = min(n0 + 16 + r16, N - 1);
    const __bf16* Ap0 = A + (size_t)ar0 * lda + half * 16;
    const __bf16* Ap1 = A + (size_t)ar1 * lda + half * 16;
    const __bf16* Bp0 = W + (size_t)br0 * ldw + half * 16;
    const __bf16* Bp1 = W + (size_t)br1 * ldw + half * 16;

    v8f acc[2][2] = {};
    for (int k0 = 0; k0 < K; k0 += 32) {
        v16bf a0 = *(const v16bf*)(Ap0 + k0);
        v16bf a1 = *(const v16bf*)(Ap1 + k0);
        v16bf b0 = *(const v16bf*)(Bp0 + k0);
        v16bf b1 = *(const v16bf*)(Bp1 + k0);
        acc[0][0] = wmma_bf16(a0, b0, acc[0][0]);
        acc[0][1] = wmma_bf16(a0, b1, acc[0][1]);
        acc[1][0] = wmma_bf16(a1, b0, acc[1][0]);
        acc[1][1] = wmma_bf16(a1, b1, acc[1][1]);
    }

#pragma unroll
    for (int im = 0; im < 2; ++im) {
#pragma unroll
        for (int in = 0; in < 2; ++in) {
            int n = n0 + in * 16 + r16;
            if (n >= N) continue;
            float bv = bias ? bias[n] : 0.f;
#pragma unroll
            for (int r = 0; r < 8; ++r) {
                int m = m0 + im * 16 + r + 8 * half;
                if (m >= M) continue;
                float v = acc[im][in][r] + bv;
                if (resid) v += resid[(size_t)m * ldc + n];
                if (act == 1) v = geluf(v);
                if (transposed) {            // bf16 store into [B][N][Lseq]
                    int bi = m / Lseq, mm = m - bi * Lseq;
                    outB[((size_t)bi * N + n) * Lseq + mm] = (__bf16)v;
                } else if (outB) {
                    outB[(size_t)m * ldc + n] = (__bf16)v;
                } else {
                    outF[(size_t)m * ldc + n] = v;
                }
            }
        }
    }
}

// ---------------- depthwise causal conv (width 4) + bias + SiLU ----------------
__global__ void conv_silu_kernel(const float* __restrict__ zx, const float* __restrict__ cw,
                                 const float* __restrict__ cb, float* __restrict__ out) {
    int idx = blockIdx.x * blockDim.x + threadIdx.x;
    if (idx >= BL * CONVD) return;
    int c  = idx % CONVD;
    int bt = idx / CONVD;
    int b  = bt >> 11;          // /LL
    int t  = bt & (LL - 1);
    float acc = cb[c];
#pragma unroll
    for (int k = 0; k < 4; ++k) {
        int tt = t + k - 3;
        if (tt >= 0) acc += cw[c * 4 + k] * zx[((size_t)b * LL + tt) * DIP + DI + c];
    }
    out[idx] = siluf(acc);
}

// ---------------- dt = softplus(.), dA = exp(-exp(A_log)*dt) ----------------
__global__ void dt_kernel(const float* __restrict__ zx, const float* __restrict__ dtb,
                          const float* __restrict__ alog, float* __restrict__ dt,
                          float* __restrict__ dA) {
    int idx = blockIdx.x * blockDim.x + threadIdx.x;
    if (idx >= BL * NH) return;
    int h = idx & (NH - 1);
    float raw = zx[(size_t)(idx >> 3) * DIP + (DIP - NH) + h] + dtb[h];
    float sp = raw > 20.f ? raw : log1pf(__expf(raw));
    dt[idx] = sp;
    dA[idx] = __expf(-__expf(alog[h]) * sp);
}

// ---------------- selective scan: one block per (b,h); 256 thr; 32 states/thr ------
// B/C strip (1KB, contiguous) for step t+1 is staged into LDS with the CDNA5 async
// LDS-DMA path (global_load_async_to_lds_b128, ASYNCcnt) while step t computes.
__global__ void scan_kernel(const float* __restrict__ xc, const float* __restrict__ dt,
                            const float* __restrict__ dA, const float* __restrict__ Dp,
                            float* __restrict__ y) {
    int b = blockIdx.x >> 3, h = blockIdx.x & 7;
    int t = threadIdx.x;
    int p = t >> 2, q = t & 3, n0 = q * 32;
    __shared__ __align__(16) float BC[2][2 * DS];   // [buf][ B(128) | C(128) ]
    float hreg[32];
#pragma unroll
    for (int j = 0; j < 32; ++j) hreg[j] = 0.f;
    float Dv = Dp[h];

    // preload step 0: 64 lanes x 16B = 1024B (B then C, contiguous in xconv row)
    if (t < 64) {
        const char* g = (const char*)(xc + ((size_t)b * LL) * CONVD + DI) + t * 16;
        unsigned l = (unsigned)(uintptr_t)((char*)&BC[0][0] + t * 16);
        asm volatile("global_load_async_to_lds_b128 %0, %1, off" :: "v"(l), "v"(g) : "memory");
    }
    asm volatile("s_wait_asynccnt 0x0" ::: "memory");
    __syncthreads();

    for (int step = 0; step < LL; ++step) {
        int cur = step & 1;
        // kick off async DMA of next step's B/C strip into the other buffer
        if (t < 64 && step + 1 < LL) {
            const char* g = (const char*)(xc + ((size_t)b * LL + step + 1) * CONVD + DI) + t * 16;
            unsigned l = (unsigned)(uintptr_t)((char*)&BC[cur ^ 1][0] + t * 16);
            asm volatile("global_load_async_to_lds_b128 %0, %1, off" :: "v"(l), "v"(g) : "memory");
        }
        const float* xr = xc + ((size_t)b * LL + step) * CONVD;
        if (step + 8 < LL)
            __builtin_prefetch(xc + ((size_t)b * LL + step + 8) * CONVD + h * HD + p, 0, 0);
        float dAv = dA[((size_t)b * LL + step) * NH + h];
        float dtv = dt[((size_t)b * LL + step) * NH + h];
        float xv  = xr[h * HD + p];
        float dtx = dtv * xv;
        const float* Bs = &BC[cur][0];
        const float* Cs = &BC[cur][DS];
        float acc = 0.f;
#pragma unroll
        for (int j = 0; j < 32; ++j) {
            hreg[j] = hreg[j] * dAv + dtx * Bs[n0 + j];
            acc += hreg[j] * Cs[n0 + j];
        }
        acc += __shfl_xor(acc, 1, 4);
        acc += __shfl_xor(acc, 2, 4);
        if (q == 0) y[((size_t)b * LL + step) * DI + h * HD + p] = acc + Dv * xv;
        // next buffer must be fully in LDS before anyone reads it next iteration
        asm volatile("s_wait_asynccnt 0x0" ::: "memory");
        __syncthreads();
    }
}

// ---------------- gated RMSNorm: out = bf16( y*silu(z) * rsqrt(mean+eps) * g ) -----
__global__ void rmsgate_kernel(const float* __restrict__ y, const float* __restrict__ zx,
                               const float* __restrict__ g, __bf16* __restrict__ out) {
    int row = blockIdx.x, t = threadIdx.x;           // 256 thr, 2 elems each
    const float* yr = y + (size_t)row * DI;
    const float* zr = zx + (size_t)row * DIP;
    float a = yr[t]       * siluf(zr[t]);
    float b = yr[t + 256] * siluf(zr[t + 256]);
    __shared__ float red[256];
    red[t] = a * a + b * b;
    __syncthreads();
    for (int s = 128; s >= 1; s >>= 1) { if (t < s) red[t] += red[t + s]; __syncthreads(); }
    float inv = rsqrtf(red[0] / (float)DI + 1e-5f);
    out[(size_t)row * DI + t]       = (__bf16)(a * inv * g[t]);
    out[(size_t)row * DI + t + 256] = (__bf16)(b * inv * g[t + 256]);
}

// ---------------- flash-style attention, WMMA scores + WMMA P@V ----------------
// block = 128 thr (4 waves), one block per (b, head, 16-row q tile)
__global__ void attn_kernel(const __bf16* __restrict__ hq, const __bf16* __restrict__ kk,
                            const __bf16* __restrict__ vT, __bf16* __restrict__ o) {
    extern __shared__ char smem[];
    float*  sc      = (float*)smem;                                  // [16][2048] f32
    __bf16* pb      = (__bf16*)(smem + 16 * LL * 4);                 // [16][2048] bf16
    float*  red     = (float*)(smem + 16 * LL * 4 + 16 * LL * 2);    // [16][8]
    float*  rowmax  = red + 128;                                     // [16]
    float*  rowsum  = rowmax + 16;                                   // [16]
    float*  part    = rowsum + 16;                                   // [4][16][32]

    int bid = blockIdx.x;
    int mt = bid & (LL / 16 - 1);
    int h  = (bid >> 7) & 7;
    int b  = bid >> 10;
    int w = threadIdx.x >> 5, lane = threadIdx.x & 31;
    int half = lane >> 4, r16 = lane & 15;
    const float scale = 0.17677669529663687f;     // 1/sqrt(32)

    v16bf aq = *(const v16bf*)(hq + ((size_t)b * LL + mt * 16 + r16) * DM + h * AHD + half * 16);

    for (int nt = w; nt < LL / 16; nt += 4) {
        v16bf bk = *(const v16bf*)(kk + ((size_t)b * LL + nt * 16 + r16) * DM + h * AHD + half * 16);
        v8f c = {};
        c = wmma_bf16(aq, bk, c);
#pragma unroll
        for (int r = 0; r < 8; ++r)
            sc[(r + 8 * half) * LL + nt * 16 + r16] = c[r] * scale;
    }
    __syncthreads();

    // row-wise max
    int row = threadIdx.x & 15, seg = threadIdx.x >> 4;   // 8 segments/row
    float mx = -1e30f;
    for (int j = seg; j < LL; j += 8) mx = fmaxf(mx, sc[row * LL + j]);
    red[row * 8 + seg] = mx;
    __syncthreads();
    if (threadIdx.x < 16) {
        float m2 = red[threadIdx.x * 8];
        for (int s = 1; s < 8; ++s) m2 = fmaxf(m2, red[threadIdx.x * 8 + s]);
        rowmax[threadIdx.x] = m2;
    }
    __syncthreads();
    // exp + row sum, probs as bf16
    float rm = rowmax[row], sum = 0.f;
    for (int j = seg; j < LL; j += 8) {
        float e = __expf(sc[row * LL + j] - rm);
        pb[row * LL + j] = (__bf16)e;
        sum += e;
    }
    red[row * 8 + seg] = sum;
    __syncthreads();
    if (threadIdx.x < 16) {
        float s2 = 0.f;
        for (int s = 0; s < 8; ++s) s2 += red[threadIdx.x * 8 + s];
        rowsum[threadIdx.x] = s2;
    }
    __syncthreads();

    // P @ V  (V pre-transposed: [B][DM][LL])
    v8f c0 = {}, c1 = {};
    const __bf16* vb = vT + (size_t)b * DM * LL + (size_t)(h * AHD) * LL;
    for (int kc = w * 32; kc < LL; kc += 128) {
        v16bf ap = *(const v16bf*)(pb + (size_t)r16 * LL + kc + half * 16);
        v16bf b0 = *(const v16bf*)(vb + (size_t)r16 * LL + kc + half * 16);
        v16bf b1 = *(const v16bf*)(vb + (size_t)(16 + r16) * LL + kc + half * 16);
        c0 = wmma_bf16(ap, b0, c0);
        c1 = wmma_bf16(ap, b1, c1);
    }
#pragma unroll
    for (int r = 0; r < 8; ++r) {
        int rr = r + 8 * half;
        part[(w * 16 + rr) * 32 + r16]      = c0[r];
        part[(w * 16 + rr) * 32 + 16 + r16] = c1[r];
    }
    __syncthreads();
    for (int idx = threadIdx.x; idx < 512; idx += 128) {
        int rr = idx >> 5, col = idx & 31;
        float v = 0.f;
#pragma unroll
        for (int ww = 0; ww < 4; ++ww) v += part[(ww * 16 + rr) * 32 + col];
        v /= rowsum[rr];
        o[((size_t)b * LL + mt * 16 + rr) * DM + h * AHD + col] = (__bf16)v;
    }
}

// ---------------- LayerNorm over 1024 -> bf16 ----------------
__global__ void ln_kernel(const float* __restrict__ x, const float* __restrict__ g,
                          const float* __restrict__ bta, __bf16* __restrict__ out) {
    int row = blockIdx.x, t = threadIdx.x;   // 256 thr, 4 elems each
    const float* xr = x + (size_t)row * MLPH;
    float v[4], s = 0.f;
#pragma unroll
    for (int j = 0; j < 4; ++j) { v[j] = xr[t + 256 * j]; s += v[j]; }
    __shared__ float red[256];
    red[t] = s; __syncthreads();
    for (int st = 128; st >= 1; st >>= 1) { if (t < st) red[t] += red[t + st]; __syncthreads(); }
    float mu = red[0] / (float)MLPH;
    __syncthreads();
    float s2 = 0.f;
#pragma unroll
    for (int j = 0; j < 4; ++j) { float d = v[j] - mu; s2 += d * d; }
    red[t] = s2; __syncthreads();
    for (int st = 128; st >= 1; st >>= 1) { if (t < st) red[t] += red[t + st]; __syncthreads(); }
    float inv = rsqrtf(red[0] / (float)MLPH + 1e-5f);
#pragma unroll
    for (int j = 0; j < 4; ++j) {
        int c = t + 256 * j;
        out[(size_t)row * MLPH + c] = (__bf16)((v[j] - mu) * inv * g[c] + bta[c]);
    }
}

// ---------------- mean over L then 2-class head ----------------
__global__ void head_kernel(const float* __restrict__ a, const float* __restrict__ hw,
                            const float* __restrict__ hb, float* __restrict__ out) {
    int b = blockIdx.x, t = threadIdx.x;     // 256 thr = model dim
    const float* ar = a + (size_t)b * LL * DM;
    float s = 0.f;
    for (int tp = 0; tp < LL; ++tp) s += ar[(size_t)tp * DM + t];
    s *= (1.f / (float)LL);
    __shared__ float red[256];
    for (int c = 0; c < 2; ++c) {
        red[t] = s * hw[c * DM + t];
        __syncthreads();
        for (int st = 128; st >= 1; st >>= 1) { if (t < st) red[t] += red[t + st]; __syncthreads(); }
        if (t == 0) out[b * 2 + c] = red[0] + hb[c];
        __syncthreads();
    }
}

// ---------------- host side ----------------
static void launch_cast(const float* s, __bf16* d, int n, hipStream_t st) {
    cast_bf16_kernel<<<(n + 255) / 256, 256, 0, st>>>(s, d, n);
}
static void launch_gemm(const __bf16* A, int lda, const __bf16* W, int ldw,
                        float* outF, __bf16* outB, int ldc, int M, int N, int K,
                        const float* bias, const float* resid, int act, int tr, int Ls,
                        hipStream_t st) {
    dim3 grid((M + 127) / 128, (N + 63) / 64);
    gemm_wmma_kernel<<<grid, 256, 0, st>>>(A, lda, W, ldw, outF, outB, ldc, M, N, K,
                                           bias, resid, act, tr, Ls);
}

extern "C" void kernel_launch(void* const* d_in, const int* in_sizes, int n_in,
                              void* d_out, int out_size, void* d_ws, size_t ws_size,
                              hipStream_t stream) {
    (void)in_sizes; (void)n_in; (void)out_size; (void)ws_size;
    const float* x        = (const float*)d_in[0];
    const float* ctx      = (const float*)d_in[1];
    const float* in_projw = (const float*)d_in[2];
    const float* conv_w   = (const float*)d_in[3];
    const float* conv_b   = (const float*)d_in[4];
    const float* dt_bias  = (const float*)d_in[5];
    const float* A_log    = (const float*)d_in[6];
    const float* D_param  = (const float*)d_in[7];
    const float* norm_w   = (const float*)d_in[8];
    const float* out_projw= (const float*)d_in[9];
    const float* wq       = (const float*)d_in[10];
    const float* wk       = (const float*)d_in[11];
    const float* wv       = (const float*)d_in[12];
    const float* wo       = (const float*)d_in[13];
    const float* wo_b     = (const float*)d_in[14];
    const float* w1       = (const float*)d_in[15];
    const float* b1       = (const float*)d_in[16];
    const float* ln_w     = (const float*)d_in[17];
    const float* ln_b     = (const float*)d_in[18];
    const float* w2       = (const float*)d_in[19];
    const float* b2       = (const float*)d_in[20];
    const float* head_w   = (const float*)d_in[21];
    const float* head_b   = (const float*)d_in[22];
    float* out = (float*)d_out;

    char* ws = (char*)d_ws;
    size_t off = 0;
    auto alloc = [&](size_t bytes) -> void* {
        size_t o = (off + 255) & ~(size_t)255;
        off = o + bytes;
        return (void*)(ws + o);
    };

    __bf16* xb    = (__bf16*)alloc((size_t)BL * DM * 2);
    __bf16* cb    = (__bf16*)alloc((size_t)BL * DM * 2);
    __bf16* wipb  = (__bf16*)alloc((size_t)DIP * DM * 2);
    __bf16* wopb  = (__bf16*)alloc((size_t)DM * DI * 2);
    __bf16* wqb   = (__bf16*)alloc((size_t)DM * DM * 2);
    __bf16* wkb   = (__bf16*)alloc((size_t)DM * DM * 2);
    __bf16* wvb   = (__bf16*)alloc((size_t)DM * DM * 2);
    __bf16* wob   = (__bf16*)alloc((size_t)DM * DM * 2);
    __bf16* w1b   = (__bf16*)alloc((size_t)MLPH * DM * 2);
    __bf16* w2b   = (__bf16*)alloc((size_t)DM * MLPH * 2);
    float*  zx    = (float*)alloc((size_t)BL * DIP * 4);
    float*  xconv = (float*)alloc((size_t)BL * CONVD * 4);
    float*  dt    = (float*)alloc((size_t)BL * NH * 4);
    float*  dA    = (float*)alloc((size_t)BL * NH * 4);
    float*  yscan = (float*)alloc((size_t)BL * DI * 4);
    __bf16* ynorm = (__bf16*)alloc((size_t)BL * DI * 2);
    float*  hres  = (float*)alloc((size_t)BL * DM * 4);
    __bf16* hb    = (__bf16*)alloc((size_t)BL * DM * 2);
    __bf16* qb    = (__bf16*)alloc((size_t)BL * DM * 2);
    __bf16* kb    = (__bf16*)alloc((size_t)BL * DM * 2);
    __bf16* vTb   = (__bf16*)alloc((size_t)BB * DM * LL * 2);
    __bf16* ob    = (__bf16*)alloc((size_t)BL * DM * 2);
    __bf16* o2b   = (__bf16*)alloc((size_t)BL * DM * 2);
    float*  hmid  = (float*)alloc((size_t)BL * MLPH * 4);
    __bf16* hln   = (__bf16*)alloc((size_t)BL * MLPH * 2);
    float*  aout  = (float*)alloc((size_t)BL * DM * 4);

    // casts
    launch_cast(x, xb, BL * DM, stream);
    launch_cast(ctx, cb, BL * DM, stream);
    launch_cast(in_projw, wipb, DIP * DM, stream);
    launch_cast(out_projw, wopb, DM * DI, stream);
    launch_cast(wq, wqb, DM * DM, stream);
    launch_cast(wk, wkb, DM * DM, stream);
    launch_cast(wv, wvb, DM * DM, stream);
    launch_cast(wo, wob, DM * DM, stream);
    launch_cast(w1, w1b, MLPH * DM, stream);
    launch_cast(w2, w2b, DM * MLPH, stream);

    // Mamba2
    launch_gemm(xb, DM, wipb, DM, zx, nullptr, DIP, BL, DIP, DM, nullptr, nullptr, 0, 0, LL, stream);
    conv_silu_kernel<<<(BL * CONVD + 255) / 256, 256, 0, stream>>>(zx, conv_w, conv_b, xconv);
    dt_kernel<<<(BL * NH + 255) / 256, 256, 0, stream>>>(zx, dt_bias, A_log, dt, dA);
    scan_kernel<<<BB * NH, 256, 0, stream>>>(xconv, dt, dA, D_param, yscan);
    rmsgate_kernel<<<BL, 256, 0, stream>>>(yscan, zx, norm_w, ynorm);
    launch_gemm(ynorm, DI, wopb, DI, hres, nullptr, DM, BL, DM, DI, nullptr, x, 0, 0, LL, stream);
    launch_cast(hres, hb, BL * DM, stream);

    // attention projections
    launch_gemm(hb, DM, wqb, DM, nullptr, qb, DM, BL, DM, DM, nullptr, nullptr, 0, 0, LL, stream);
    launch_gemm(cb, DM, wkb, DM, nullptr, kb, DM, BL, DM, DM, nullptr, nullptr, 0, 0, LL, stream);
    launch_gemm(cb, DM, wvb, DM, nullptr, vTb, DM, BL, DM, DM, nullptr, nullptr, 0, 1, LL, stream);

    // attention core
    size_t shmem = (size_t)16 * LL * 4 + (size_t)16 * LL * 2 + (128 + 16 + 16 + 4 * 16 * 32) * 4;
    attn_kernel<<<BB * AH * (LL / 16), 128, shmem, stream>>>(hb, kb, vTb, ob);

    // out proj + MLP
    launch_gemm(ob, DM, wob, DM, nullptr, o2b, DM, BL, DM, DM, wo_b, nullptr, 0, 0, LL, stream);
    launch_gemm(o2b, DM, w1b, DM, hmid, nullptr, MLPH, BL, MLPH, DM, b1, nullptr, 1, 0, LL, stream);
    ln_kernel<<<BL, 256, 0, stream>>>(hmid, ln_w, ln_b, hln);
    launch_gemm(hln, MLPH, w2b, MLPH, aout, nullptr, DM, BL, DM, MLPH, b2, nullptr, 0, 0, LL, stream);

    // mean + classifier head
    head_kernel<<<BB, 256, 0, stream>>>(aout, head_w, head_b, out);
}
